// FittedWarpOld_38027640439464
// MI455X (gfx1250) — matrix operand order repaired and checked by reading, not compiled, not run on hardware
//
#include <hip/hip_runtime.h>
#include <hip/hip_bf16.h>
#include <math.h>

typedef __attribute__((ext_vector_type(16))) _Float16 v16h;
typedef __attribute__((ext_vector_type(8)))  float    v8f;
typedef __attribute__((ext_vector_type(4)))  float    fvec4;

#define N_ROWS 8192
#define D_COLS 1024

// ---------------------------------------------------------------------------
// Kernel 1: a[i] = sigmoid(dot(x[i,:], w))  via v_wmma_f32_16x16x32_f16.
// Each wave32 owns a 16-row tile; K loop steps 64 with two independent
// accumulators (no WMMA->WMMA C-chain per iteration, fewer hazard NOPs).
// B broadcasts w across all 16 N-columns: B[K][N] = w[k+K] for every N, so
// every lane loads identical (cache-broadcast) addresses -- no exec-mask
// divergence in the hot loop -- and every column of D holds the dot product.
// ---------------------------------------------------------------------------
__global__ __launch_bounds__(256) void gemv_sigmoid_wmma(
    const float* __restrict__ x, const float* __restrict__ w,
    float* __restrict__ a)
{
  const int lane = threadIdx.x & 31;
  const int wave = threadIdx.x >> 5;
  const int tile = blockIdx.x * 8 + wave;   // one 16-row tile per wave
  const int rowBase = tile * 16;
  const int m = lane & 15;                  // row within tile (A) / N column (B,D)
  const int h = lane >> 4;                  // lane half selects K sub-block
  const float* xrow = x + (size_t)(rowBase + m) * D_COLS;

  v8f acc0 = {}, acc1 = {};
  for (int k = 0; k < D_COLS; k += 64) {
    const float* pa = xrow + k;
    const float* pw = w + k;
    // A tiles (16x32 f16 each), CDNA5 16-bit A layout:
    //   lane-half h: elems 0..7 <- K = h*8+0..7 ; elems 8..15 <- K = 16+h*8+0..7
    v16h A0, A1;
    #pragma unroll
    for (int j = 0; j < 8; ++j) {
      A0[j]     = (_Float16)pa[h * 8 + j];
      A0[8 + j] = (_Float16)pa[16 + h * 8 + j];
      A1[j]     = (_Float16)pa[32 + h * 8 + j];
      A1[8 + j] = (_Float16)pa[48 + h * 8 + j];
    }
    // B tiles (32x16 f16): lane-half h: elems 0..7 <- K = h*8+0..7 ; elems 8..15 <- K = 16+h*8+0..7;
    // value depends only on K (broadcast across N).
    v16h B0, B1;
    #pragma unroll
    for (int j = 0; j < 8; ++j) {
      B0[j]     = (_Float16)pw[h * 8 + j];
      B0[8 + j] = (_Float16)pw[16 + h * 8 + j];
      B1[j]     = (_Float16)pw[32 + h * 8 + j];
      B1[8 + j] = (_Float16)pw[48 + h * 8 + j];
    }
    acc0 = __builtin_amdgcn_wmma_f32_16x16x32_f16(
        false, A0, false, B0, (short)0, acc0, false, false);
    acc1 = __builtin_amdgcn_wmma_f32_16x16x32_f16(
        false, A1, false, B1, (short)0, acc1, false, false);
  }
  // D layout: VGPR j, lanes 0-15 -> (M=j, N=lane); lanes 16-31 -> (M=8+j).
  // Every N column holds the dot, so one lane per half writes 8 rows.
  if (m == 0) {
    #pragma unroll
    for (int j = 0; j < 8; ++j) {
      float z = acc0[j] + acc1[j];
      a[rowBase + h * 8 + j] = 1.0f / (1.0f + __expf(-z));
    }
  }
}

// ---------------------------------------------------------------------------
// Kernel 2: single-block scan over a[8192]; emits per-column weights and the
// segment-start table rowstart[] (rowstart[r] = first i with idx_i == r).
// Since a_i < 1, idx is monotone with steps in {0,1}; every bin 0..idx_max
// is hit, and rows beyond idx_max gather an empty span -> exact zeros.
// ---------------------------------------------------------------------------
__global__ __launch_bounds__(1024) void scan_bins(
    const float* __restrict__ a, float* __restrict__ mainw,
    float* __restrict__ crossw, int* __restrict__ rowstart)
{
  __shared__ float sm[1024];
  const int t = threadIdx.x;

  // init rowstart to N so un-written rows produce empty spans / no cross term
  for (int r = t; r <= N_ROWS; r += 1024) rowstart[r] = N_ROWS;
  __threadfence();

  float v[8];
  float s = 0.f;
  #pragma unroll
  for (int j = 0; j < 8; ++j) { v[j] = a[t * 8 + j]; s += v[j]; }
  sm[t] = s;
  __syncthreads();

  // Hillis-Steele inclusive scan over the 1024 per-thread partials
  for (int off = 1; off < 1024; off <<= 1) {
    float add = (t >= off) ? sm[t - off] : 0.f;
    __syncthreads();
    sm[t] += add;
    __syncthreads();
  }

  float excl = sm[t] - s;               // b_{8t-1}; 0 for t==0
  float run  = excl;
  int prev   = (int)floorf(excl);       // == 0 at i==0, matching reference prev[0]=0
  if (t == 0) rowstart[0] = 0;
  #pragma unroll
  for (int j = 0; j < 8; ++j) {
    const int i = t * 8 + j;
    run += v[j];                        // b_i
    int idx    = (int)floorf(run);
    float frac = run - (float)idx;
    bool same  = (idx == prev);
    mainw[i]  = same ? v[j] : frac;
    crossw[i] = same ? 0.f  : (v[j] - frac);
    if (!same) rowstart[idx] = i;       // unique writer per bin
    prev = idx;
  }
}

// ---------------------------------------------------------------------------
// Kernel 3: out[r,:] = sum_{i in [s_r, s_{r+1})} main_i * x[i,:]
//                    + cross_{s_{r+1}} * x[s_{r+1},:]   (if s_{r+1} < N)
// One block per output row; 256 threads x float4 = 1024 columns. Atomic-free,
// fully coalesced; writes every output element (zeros for empty rows).
// ---------------------------------------------------------------------------
__global__ __launch_bounds__(256) void bin_gather(
    const float* __restrict__ x, const float* __restrict__ mainw,
    const float* __restrict__ crossw, const int* __restrict__ rowstart,
    float* __restrict__ out)
{
  const int r = blockIdx.x;
  const int c = threadIdx.x * 4;
  const int s = rowstart[r];
  const int e = rowstart[r + 1];

  fvec4 acc = {0.f, 0.f, 0.f, 0.f};
  for (int i = s; i < e; ++i) {
    if (i + 1 < N_ROWS)
      __builtin_prefetch(x + (size_t)(i + 1) * D_COLS + c, 0, 1);
    const float wgt = mainw[i];
    fvec4 xv = *(const fvec4*)(x + (size_t)i * D_COLS + c);
    acc += wgt * xv;
  }
  if (e < N_ROWS) {
    const float wgt = crossw[e];
    fvec4 xv = *(const fvec4*)(x + (size_t)e * D_COLS + c);
    acc += wgt * xv;
  }
  *(fvec4*)(out + (size_t)r * D_COLS + c) = acc;
}

// ---------------------------------------------------------------------------
extern "C" void kernel_launch(void* const* d_in, const int* in_sizes, int n_in,
                              void* d_out, int out_size, void* d_ws, size_t ws_size,
                              hipStream_t stream) {
  const float* x = (const float*)d_in[0];   // [8192, 1024] f32
  const float* w = (const float*)d_in[1];   // [1024, 1]    f32
  float* out = (float*)d_out;               // [8192, 1024] f32

  // workspace layout: a | main | cross | rowstart  (~128 KB total)
  float* a        = (float*)d_ws;
  float* mainw    = a + N_ROWS;
  float* crossw   = mainw + N_ROWS;
  int*   rowstart = (int*)(crossw + N_ROWS);   // N_ROWS + 1 entries

  // K1: 64 blocks x 8 waves x 16 rows = 8192 rows
  gemv_sigmoid_wmma<<<64, 256, 0, stream>>>(x, w, a);
  // K2: single-workgroup scan (8192 elements)
  scan_bins<<<1, 1024, 0, stream>>>(a, mainw, crossw, rowstart);
  // K3: one block per output row
  bin_gather<<<N_ROWS, 256, 0, stream>>>(x, mainw, crossw, rowstart, out);
  (void)in_sizes; (void)n_in; (void)out_size; (void)ws_size;
}
